// ArcFaceLoss_17154099380512
// MI455X (gfx1250) — matrix operand order repaired and checked by reading, not compiled, not run on hardware
//
#include <hip/hip_runtime.h>

#define B_ROWS   1024
#define EMB_D    512
#define NCLS     100000
#define NT       64                 // classes per tile
#define NSUB     (NT / 16)          // 4 WMMA column subtiles
#define NTILES   ((NCLS + NT - 1) / NT)   // 1563 (last tile ragged: 32 valid)
#define SCALE_F  30.0f
#define MARGIN_F 0.5f
#define NEG_BIG  (-1.0e30f)
#define LDSROW   (EMB_D + 8)        // padded bf16 row (bank-conflict avoidance)

typedef __bf16 bf16_t;
typedef __attribute__((ext_vector_type(16))) __bf16 v16bf;
typedef __attribute__((ext_vector_type(8)))  __bf16 v8bf;
typedef __attribute__((ext_vector_type(4)))  __bf16 v4bf;
typedef __attribute__((ext_vector_type(8)))  float  v8f;

union V16U { v16bf v; v8bf h[2]; };

__device__ __forceinline__ bf16_t f2hi(float x) { return (bf16_t)x; }
__device__ __forceinline__ bf16_t f2lo(float x) {
  float hf = (float)((bf16_t)x);
  return (bf16_t)(x - hf);
}

// ---------------- Kernel 0: split embeddings into bf16 hi/lo ----------------
__global__ void split_emb_kernel(const float* __restrict__ emb,
                                 bf16_t* __restrict__ ehi,
                                 bf16_t* __restrict__ elo) {
  int i = blockIdx.x * blockDim.x + threadIdx.x;   // one float4 (131072 total)
  float4 v = ((const float4*)emb)[i];
  v4bf h, l;
  h[0] = f2hi(v.x); l[0] = f2lo(v.x);
  h[1] = f2hi(v.y); l[1] = f2lo(v.y);
  h[2] = f2hi(v.z); l[2] = f2lo(v.z);
  h[3] = f2hi(v.w); l[3] = f2lo(v.w);
  *(v4bf*)(ehi + 4 * (size_t)i) = h;
  *(v4bf*)(elo + 4 * (size_t)i) = l;
}

// --------- Kernel 1: fused norm + GEMM (bf16x3 WMMA) + online softmax -------
__global__ __launch_bounds__(512) void
arcface_gemm_kernel(const float* __restrict__ W,
                    const int*   __restrict__ labels,
                    const bf16_t* __restrict__ ehi,
                    const bf16_t* __restrict__ elo,
                    float* __restrict__ part_max,
                    float* __restrict__ part_sum,
                    float* __restrict__ label_logit) {
  // 64*520*2 bf16 * 2 arrays = 133,120 B + norms
  __shared__ __align__(16) unsigned char smem[NT * LDSROW * 2 * 2 + 512];
  bf16_t* sWhi  = (bf16_t*)smem;
  bf16_t* sWlo  = sWhi + NT * LDSROW;
  float*  sNorm = (float*)(smem + NT * LDSROW * 2 * 2);

  const int t     = threadIdx.x;
  const int tile  = blockIdx.x;
  const int cbase = tile * NT;
  const int wave  = t >> 5;
  const int lane  = t & 31;

  // ---- load 64x512 f32 W tile, split to bf16 hi/lo in LDS, row norms ----
  // 16 waves x 4 rows each; each row: 32 lanes x 16 floats (4 float4)
  {
#pragma unroll
    for (int r = 0; r < 4; ++r) {
      const int cr = wave * 4 + r;
      int grow = cbase + cr;
      if (grow > NCLS - 1) grow = NCLS - 1;      // clamp ragged last tile
      const float4* wr = (const float4*)(W + (size_t)grow * EMB_D + lane * 16);
      float ssq = 0.f;
#pragma unroll
      for (int i = 0; i < 4; ++i) {
        float4 w4 = wr[i];
        int k = lane * 16 + i * 4;
        ssq += w4.x * w4.x + w4.y * w4.y + w4.z * w4.z + w4.w * w4.w;
        v4bf h, l;
        h[0] = f2hi(w4.x); l[0] = f2lo(w4.x);
        h[1] = f2hi(w4.y); l[1] = f2lo(w4.y);
        h[2] = f2hi(w4.z); l[2] = f2lo(w4.z);
        h[3] = f2hi(w4.w); l[3] = f2lo(w4.w);
        *(v4bf*)(sWhi + cr * LDSROW + k) = h;
        *(v4bf*)(sWlo + cr * LDSROW + k) = l;
      }
#pragma unroll
      for (int off = 16; off >= 1; off >>= 1) ssq += __shfl_xor(ssq, off);
      if (lane == 0) sNorm[cr] = fmaxf(sqrtf(ssq), 1e-12f);
    }
  }
  __syncthreads();

  // ---- GEMM: 16 waves x (64 rows x 64 cols), K=512 in 16 steps of 32 ----
  const int lrow    = lane & 15;
  const int halfsel = lane >> 4;

  v8f acc[4][NSUB] = {};

  // B fragment (32x16): lane<16 -> col lrow, K {k0..+7, k0+16..+23}; lane>=16 -> K {k0+8..+15, k0+24..+31}
  const bf16_t* sBhi = sWhi + lrow * LDSROW + halfsel * 8;
  const bf16_t* sBlo = sWlo + lrow * LDSROW + halfsel * 8;
  // A fragment (16x32): lane<16 -> row lrow, K {k0..+7, k0+16..+23};
  //                     lane>=16 ->           K {k0+8..+15, k0+24..+31}
  const bf16_t* gahi = ehi + (size_t)(wave * 64 + lrow) * EMB_D + halfsel * 8;
  const bf16_t* galo = elo + (size_t)(wave * 64 + lrow) * EMB_D + halfsel * 8;

  for (int k0 = 0; k0 < EMB_D; k0 += 32) {
    // keep all 4 hi/lo B fragments resident for this K step (64 VGPRs)
    V16U bh[NSUB], bl[NSUB];
#pragma unroll
    for (int n = 0; n < NSUB; ++n) {
      const bf16_t* ph = sBhi + n * 16 * LDSROW + k0;
      const bf16_t* pl = sBlo + n * 16 * LDSROW + k0;
      bh[n].h[0] = *(const v8bf*)(ph);
      bh[n].h[1] = *(const v8bf*)(ph + 16);
      bl[n].h[0] = *(const v8bf*)(pl);
      bl[n].h[1] = *(const v8bf*)(pl + 16);
    }
#pragma unroll
    for (int m = 0; m < 4; ++m) {
      const bf16_t* pah = gahi + (size_t)m * 16 * EMB_D + k0;
      const bf16_t* pal = galo + (size_t)m * 16 * EMB_D + k0;
      V16U ah, al;
      ah.h[0] = *(const v8bf*)(pah);
      ah.h[1] = *(const v8bf*)(pah + 16);
      al.h[0] = *(const v8bf*)(pal);
      al.h[1] = *(const v8bf*)(pal + 16);
      // 12 back-to-back WMMAs per A-fragment group (bf16x3 precision scheme)
#pragma unroll
      for (int n = 0; n < NSUB; ++n) {
        acc[m][n] = __builtin_amdgcn_wmma_f32_16x16x32_bf16(
            false, ah.v, false, bh[n].v, (short)0, acc[m][n], false, false);
        acc[m][n] = __builtin_amdgcn_wmma_f32_16x16x32_bf16(
            false, ah.v, false, bl[n].v, (short)0, acc[m][n], false, false);
        acc[m][n] = __builtin_amdgcn_wmma_f32_16x16x32_bf16(
            false, al.v, false, bh[n].v, (short)0, acc[m][n], false, false);
      }
    }
  }

  __syncthreads();                       // W tile dead -> reuse LDS for staging
  float* sPM = (float*)smem;             // 1024 floats
  float* sPS = sPM + B_ROWS;             // 1024 floats (inside old sWhi region)

  // ---- epilogue: normalize, margin, scale, per-row (max,sumexp) over 64 cols
  float inv_n[NSUB];
#pragma unroll
  for (int n = 0; n < NSUB; ++n) inv_n[n] = 1.0f / sNorm[n * 16 + lrow];

#pragma unroll
  for (int m = 0; m < 4; ++m) {
#pragma unroll
    for (int r = 0; r < 8; ++r) {
      const int b   = wave * 64 + m * 16 + r + halfsel * 8;
      const int lab = labels[b];
      float lgt[NSUB];
#pragma unroll
      for (int n = 0; n < NSUB; ++n) {
        const int c = cbase + n * 16 + lrow;
        float v = acc[m][n][r] * inv_n[n];
        if (c == lab) {
          v = SCALE_F * (v - MARGIN_F);
          label_logit[b] = v;            // exactly one (b,c) match grid-wide
        } else {
          v = SCALE_F * v;
        }
        lgt[n] = (c < NCLS) ? v : NEG_BIG;   // mask ragged last tile
      }
      float mx = lgt[0];
#pragma unroll
      for (int n = 1; n < NSUB; ++n) mx = fmaxf(mx, lgt[n]);
      float s = 0.f;
#pragma unroll
      for (int n = 0; n < NSUB; ++n) s += __expf(lgt[n] - mx);
      // online softmax merge across the 16 lanes of this half (4 classes each)
#pragma unroll
      for (int off = 1; off < 16; off <<= 1) {
        float mo = __shfl_xor(mx, off);
        float so = __shfl_xor(s, off);
        float nm = fmaxf(mx, mo);
        s = s * __expf(mx - nm) + so * __expf(mo - nm);
        mx = nm;
      }
      if (lrow == 0) { sPM[b] = mx; sPS[b] = s; }
    }
  }
  __syncthreads();
  for (int i = t; i < B_ROWS; i += 512) {        // coalesced partial write
    part_max[(size_t)tile * B_ROWS + i] = sPM[i];
    part_sum[(size_t)tile * B_ROWS + i] = sPS[i];
  }
}

// -------- Kernel 2: merge per-tile (max,sumexp) -> per-row loss --------
__global__ __launch_bounds__(256) void
logsumexp_reduce_kernel(const float* __restrict__ part_max,
                        const float* __restrict__ part_sum,
                        const float* __restrict__ label_logit,
                        float* __restrict__ lossvec) {
  const int t  = threadIdx.x;      // 256 = 32 rows x 8 tile-lanes
  const int bo = t & 31;
  const int tl = t >> 5;
  const int b  = blockIdx.x * 32 + bo;
  float mx = -INFINITY, s = 0.f;
  for (int tile = tl; tile < NTILES; tile += 8) {   // coalesced in b
    float m2 = part_max[(size_t)tile * B_ROWS + b];
    float s2 = part_sum[(size_t)tile * B_ROWS + b];
    float nm = fmaxf(mx, m2);
    s  = s * expf(mx - nm) + s2 * expf(m2 - nm);
    mx = nm;
  }
  __shared__ float sm[256], ss[256];
  sm[t] = mx; ss[t] = s;
  __syncthreads();
  for (int o = 128; o >= 32; o >>= 1) {
    if (t < o) {
      float ma = sm[t], mb = sm[t + o], sa = ss[t], sb = ss[t + o];
      float nm = fmaxf(ma, mb);
      sm[t] = nm;
      ss[t] = sa * expf(ma - nm) + sb * expf(mb - nm);
    }
    __syncthreads();
  }
  if (t < 32) lossvec[b] = sm[t] + logf(ss[t]) - label_logit[b];
}

// -------- Kernel 3: mean over batch -> scalar loss --------
__global__ __launch_bounds__(256) void
final_mean_kernel(const float* __restrict__ lossvec, float* __restrict__ out) {
  __shared__ float sm[256];
  const int t = threadIdx.x;
  sm[t] = lossvec[t] + lossvec[t + 256] + lossvec[t + 512] + lossvec[t + 768];
  __syncthreads();
  for (int o = 128; o > 0; o >>= 1) {
    if (t < o) sm[t] += sm[t + o];
    __syncthreads();
  }
  if (t == 0) out[0] = sm[0] * (1.0f / (float)B_ROWS);
}

extern "C" void kernel_launch(void* const* d_in, const int* in_sizes, int n_in,
                              void* d_out, int out_size, void* d_ws, size_t ws_size,
                              hipStream_t stream) {
  const float* emb    = (const float*)d_in[0];   // [1024,512] f32
  const int*   labels = (const int*)d_in[1];     // [1024]
  const float* W      = (const float*)d_in[2];   // [100000,512] f32
  float* out = (float*)d_out;

  // workspace layout
  bf16_t* ehi = (bf16_t*)d_ws;                                   // 1 MB
  bf16_t* elo = ehi + (size_t)B_ROWS * EMB_D;                    // 1 MB
  float* part_max    = (float*)((char*)d_ws + 2u * B_ROWS * EMB_D * 2u);
  float* part_sum    = part_max + (size_t)NTILES * B_ROWS;       // 6.4 MB each
  float* label_logit = part_sum + (size_t)NTILES * B_ROWS;       // 4 KB
  float* lossvec     = label_logit + B_ROWS;                     // 4 KB

  split_emb_kernel<<<(B_ROWS * EMB_D / 4) / 256, 256, 0, stream>>>(emb, ehi, elo);
  arcface_gemm_kernel<<<NTILES, 512, 0, stream>>>(W, labels, ehi, elo,
                                                  part_max, part_sum, label_logit);
  logsumexp_reduce_kernel<<<B_ROWS / 32, 256, 0, stream>>>(part_max, part_sum,
                                                           label_logit, lossvec);
  final_mean_kernel<<<1, 256, 0, stream>>>(lossvec, out);
}
